// SwinTransBlock_47476568490601
// MI455X (gfx1250) — matrix operand order repaired and checked by reading, not compiled, not run on hardware
//
#include <hip/hip_runtime.h>
#include <hip/hip_bf16.h>

typedef __bf16 v16bf __attribute__((ext_vector_type(16)));
typedef float  v8f   __attribute__((ext_vector_type(8)));

#define SHIFT   4
#define SCALE   0.17677669529663687f   // 32^-0.5

// LDS maps (bytes)
#define ATT_LDS 182272   // X 32K | Q 32K | K 32K | V 32K | O 32K | P 16K | bias8 2K
#define MLP_LDS 163840   // Xn 32K | H 128K

union FragU { uint4 u[2]; v16bf v; };

static __device__ __forceinline__ v16bf ld_frag(const __bf16* p0, const __bf16* p1) {
  FragU f;
  f.u[0] = *(const uint4*)p0;
  f.u[1] = *(const uint4*)p1;
  return f.v;
}

static __device__ __forceinline__ float wave_sum(float v) {
  for (int m = 16; m >= 1; m >>= 1) v += __shfl_xor(v, m, 32);
  return v;
}
// reductions confined to a 16-lane half (row of a WMMA D tile)
static __device__ __forceinline__ float half_sum(float v) {
  for (int m = 8; m >= 1; m >>= 1) v += __shfl_xor(v, m, 32);
  return v;
}
static __device__ __forceinline__ float half_max(float v) {
  for (int m = 8; m >= 1; m >>= 1) v = fmaxf(v, __shfl_xor(v, m, 32));
  return v;
}

// ---------------- weight convert: dst[n][k] = (bf16) src[k][n] ----------------
__global__ void wconv_kernel(const float* __restrict__ src, __bf16* __restrict__ dst,
                             int K, int N) {
  int i = blockIdx.x * 256 + threadIdx.x;
  if (i >= K * N) return;
  int n = i / K, k = i - n * K;
  dst[i] = (__bf16)src[k * N + n];
}

// ---------------- LN1 + cyclic shift + window partition -> bf16 ----------------
__global__ void ln1_kernel(const float* __restrict__ x, const float* __restrict__ g,
                           const float* __restrict__ bta, __bf16* __restrict__ xw) {
  int tid = threadIdx.x, wave = tid >> 5, lane = tid & 31;
  int d   = blockIdx.x * 8 + wave;          // destination token id (win*64+n)
  int win = d >> 6, n = d & 63;
  int bi = win >> 6, wi = win & 63;
  int wr = wi >> 3, wc = wi & 7, ir = n >> 3, ic = n & 7;
  int hh = (wr * 8 + ir + SHIFT) & 63;
  int ww = (wc * 8 + ic + SHIFT) & 63;
  const float* src = x + ((size_t)bi * 4096 + hh * 64 + ww) * 256 + lane * 8;

  float4 a0 = *(const float4*)src, a1 = *(const float4*)(src + 4);
  float v[8] = {a0.x, a0.y, a0.z, a0.w, a1.x, a1.y, a1.z, a1.w};
  float s = 0.f, sq = 0.f;
#pragma unroll
  for (int i = 0; i < 8; ++i) { s += v[i]; sq += v[i] * v[i]; }
  s = wave_sum(s); sq = wave_sum(sq);
  float mean = s * (1.f / 256.f);
  float var  = sq * (1.f / 256.f) - mean * mean;
  float rs   = rsqrtf(var + 1e-5f);

  __attribute__((aligned(16))) __bf16 o[8];
#pragma unroll
  for (int i = 0; i < 8; ++i) {
    int c = lane * 8 + i;
    o[i] = (__bf16)((v[i] - mean) * rs * g[c] + bta[c]);
  }
  *(uint4*)(xw + (size_t)d * 256 + lane * 8) = *(const uint4*)o;
}

// ---------------- fused window attention block (one block per window) ----------------
__global__ void attn_kernel(const __bf16* __restrict__ xw,
                            const __bf16* __restrict__ wqkvT, const float* __restrict__ qkv_b,
                            const __bf16* __restrict__ wprojT, const float* __restrict__ proj_b,
                            const float* __restrict__ relb,
                            const float* __restrict__ x_in,
                            float* __restrict__ out) {
  extern __shared__ char smem[];
  __bf16* sX = (__bf16*)(smem);            // [64][256]  LN'd window activations
  __bf16* sQ = (__bf16*)(smem + 32768);    // [8][64][32] scaled q
  __bf16* sK = (__bf16*)(smem + 65536);    // [8][64][32]
  __bf16* sV = (__bf16*)(smem + 98304);    // [8][32][64] (transposed for B frags)
  __bf16* sO = (__bf16*)(smem + 131072);   // [64][256]  attention output
  __bf16* sP = (__bf16*)(smem + 163840);   // [8][16][64] probs (one row-block per head)
  float*  sB = (float*)(smem + 180224);    // [8][8][8]  relative-position bias

  int tid = threadIdx.x, wave = tid >> 5, lane = tid & 31;
  int l16 = lane & 15, lh = lane >> 4;
  int win = blockIdx.x;
  int bi = win >> 6, wi = win & 63, wr = wi >> 3, wc = wi & 7;

  // stage window activations + bias table
  {
    const uint4* src = (const uint4*)(xw + (size_t)win * (64 * 256));
    uint4* dst = (uint4*)sX;
    for (int i = tid; i < 2048; i += 256) dst[i] = src[i];
  }
  for (int i = tid; i < 512; i += 256) {               // bias8[h][ir][jr]
    int h = i >> 6, di = (i >> 3) & 7, dj = i & 7;
    sB[i] = relb[(((dj - di) + 7) * 15) * 8 + h];
  }
  __syncthreads();

  // ---- QKV GEMM: [64x256] @ [256x768] -> q,k,v in LDS ----
  for (int t = 0; t < 24; ++t) {
    int nt = wave * 6 + (t >> 2), mt = t & 3;
    v8f acc = {};
    const __bf16* xa = sX + (mt * 16 + l16) * 256 + lh * 8;
    const __bf16* wb = wqkvT + ((size_t)(nt * 16 + l16)) * 256 + lh * 8;
#pragma unroll
    for (int kt = 0; kt < 8; ++kt) {
      v16bf a  = ld_frag(xa + kt * 32, xa + kt * 32 + 16);
      v16bf bf = ld_frag(wb + kt * 32, wb + kt * 32 + 16);
      acc = __builtin_amdgcn_wmma_f32_16x16x32_bf16(false, a, false, bf, (short)0, acc, false, false);
    }
    int ng = nt * 16 + l16;
    float bias = qkv_b[ng];
#pragma unroll
    for (int r = 0; r < 8; ++r) {
      int m = mt * 16 + r + 8 * lh;
      float val = acc[r] + bias;
      if (ng < 256) {
        sQ[(ng >> 5) * 2048 + m * 32 + (ng & 31)] = (__bf16)(val * SCALE);
      } else if (ng < 512) {
        int nk = ng - 256;
        sK[(nk >> 5) * 2048 + m * 32 + (nk & 31)] = (__bf16)val;
      } else {
        int nv = ng - 512;
        sV[(nv >> 5) * 2048 + (nv & 31) * 64 + m] = (__bf16)val;   // transposed
      }
    }
  }
  __syncthreads();

  // ---- per-head attention (wave == head) ----
  int h = wave;
  const __bf16* qh = sQ + h * 2048;
  const __bf16* kh = sK + h * 2048;
  const __bf16* vh = sV + h * 2048;
  __bf16* ph = sP + h * 1024;

  for (int ti = 0; ti < 4; ++ti) {
    // scores: q[16x32] @ k^T[32x64]
    v8f sc[4] = {{}, {}, {}, {}};
    const __bf16* qa = qh + (ti * 16 + l16) * 32 + lh * 8;
    v16bf aq = ld_frag(qa, qa + 16);
#pragma unroll
    for (int tj = 0; tj < 4; ++tj) {
      const __bf16* kb = kh + (tj * 16 + l16) * 32 + lh * 8;
      v16bf bk = ld_frag(kb, kb + 16);
      sc[tj] = __builtin_amdgcn_wmma_f32_16x16x32_bf16(false, aq, false, bk, (short)0, sc[tj], false, false);
    }
    // bias + shift-mask + row softmax (rows live inside a 16-lane half)
#pragma unroll
    for (int r = 0; r < 8; ++r) {
      int i  = ti * 16 + r + 8 * lh;
      int iir = i >> 3, iic = i & 7;
      int lhi = (wr == 7) ? ((iir < 4) ? 1 : 2) : 0;
      int lwi = (wc == 7) ? ((iic < 4) ? 1 : 2) : 0;
      float sv[4];
      float mx = -3.0e38f;
#pragma unroll
      for (int tj = 0; tj < 4; ++tj) {
        int j = tj * 16 + l16;
        int jjr = j >> 3, jjc = j & 7;
        int lhj = (wr == 7) ? ((jjr < 4) ? 1 : 2) : 0;
        int lwj = (wc == 7) ? ((jjc < 4) ? 1 : 2) : 0;
        float msk = (lhi != lhj || lwi != lwj) ? -100.f : 0.f;
        float val = sc[tj][r] + sB[h * 64 + iir * 8 + jjr] + msk;
        sv[tj] = val;
        mx = fmaxf(mx, val);
      }
      mx = half_max(mx);
      float sum = 0.f;
#pragma unroll
      for (int tj = 0; tj < 4; ++tj) { sv[tj] = __expf(sv[tj] - mx); sum += sv[tj]; }
      sum = half_sum(sum);
      float inv = 1.f / sum;
      int mloc = r + 8 * lh;
#pragma unroll
      for (int tj = 0; tj < 4; ++tj)
        ph[mloc * 64 + tj * 16 + l16] = (__bf16)(sv[tj] * inv);
    }
    __syncthreads();   // probs visible as A-fragments

    // attn[16x64] @ v[64x32]
    const __bf16* pa = ph + l16 * 64;
    v16bf ap0 = ld_frag(pa + lh * 8,      pa + 16 + lh * 8);
    v16bf ap1 = ld_frag(pa + 32 + lh * 8, pa + 48 + lh * 8);
#pragma unroll
    for (int nd = 0; nd < 2; ++nd) {
      v8f oa = {};
      const __bf16* vb = vh + (nd * 16 + l16) * 64 + lh * 8;
      v16bf bv0 = ld_frag(vb,      vb + 16);
      v16bf bv1 = ld_frag(vb + 32, vb + 48);
      oa = __builtin_amdgcn_wmma_f32_16x16x32_bf16(false, ap0, false, bv0, (short)0, oa, false, false);
      oa = __builtin_amdgcn_wmma_f32_16x16x32_bf16(false, ap1, false, bv1, (short)0, oa, false, false);
#pragma unroll
      for (int r = 0; r < 8; ++r) {
        int m = ti * 16 + r + 8 * lh;
        sO[m * 256 + h * 32 + nd * 16 + l16] = (__bf16)oa[r];
      }
    }
    __syncthreads();
  }

  // ---- proj GEMM + window-reverse + un-shift + residual -> out (= x2 stream) ----
  for (int t = 0; t < 8; ++t) {
    int nt = wave * 2 + (t >> 2), mt = t & 3;
    v8f acc = {};
    const __bf16* oa = sO + (mt * 16 + l16) * 256 + lh * 8;
    const __bf16* wb = wprojT + ((size_t)(nt * 16 + l16)) * 256 + lh * 8;
#pragma unroll
    for (int kt = 0; kt < 8; ++kt) {
      v16bf a  = ld_frag(oa + kt * 32, oa + kt * 32 + 16);
      v16bf bf = ld_frag(wb + kt * 32, wb + kt * 32 + 16);
      acc = __builtin_amdgcn_wmma_f32_16x16x32_bf16(false, a, false, bf, (short)0, acc, false, false);
    }
    int c = nt * 16 + l16;
    float pb = proj_b[c];
#pragma unroll
    for (int r = 0; r < 8; ++r) {
      int m = mt * 16 + r + 8 * lh;
      int iir = m >> 3, iic = m & 7;
      int gh = (wr * 8 + iir + SHIFT) & 63;
      int gw = (wc * 8 + iic + SHIFT) & 63;
      size_t idx = ((size_t)bi * 4096 + gh * 64 + gw) * 256 + c;
      out[idx] = acc[r] + pb + x_in[idx];
    }
  }
}

// ---------------- fused MLP: LN2 -> fc1 -> GELU -> fc2 -> residual ----------------
__global__ void mlp_kernel(const float* __restrict__ g2, const float* __restrict__ b2,
                           const __bf16* __restrict__ wfc1T, const float* __restrict__ fc1_b,
                           const __bf16* __restrict__ wfc2T, const float* __restrict__ fc2_b,
                           float* __restrict__ out) {
  extern __shared__ char smem[];
  __bf16* sX = (__bf16*)smem;             // [64][256]  LN2 output
  __bf16* sH = (__bf16*)(smem + 32768);   // [64][1024] GELU(fc1) — 128KB LDS tile
  int tid = threadIdx.x, wave = tid >> 5, lane = tid & 31;
  int l16 = lane & 15, lh = lane >> 4;
  size_t t0 = (size_t)blockIdx.x * 64;

  // LN2 (wave per token, 8 tokens per wave)
  for (int tk = 0; tk < 8; ++tk) {
    int ml = wave * 8 + tk;
    const float* src = out + (t0 + ml) * 256 + lane * 8;
    float4 a0 = *(const float4*)src, a1 = *(const float4*)(src + 4);
    float v[8] = {a0.x, a0.y, a0.z, a0.w, a1.x, a1.y, a1.z, a1.w};
    float s = 0.f, sq = 0.f;
#pragma unroll
    for (int i = 0; i < 8; ++i) { s += v[i]; sq += v[i] * v[i]; }
    s = wave_sum(s); sq = wave_sum(sq);
    float mean = s * (1.f / 256.f), var = sq * (1.f / 256.f) - mean * mean;
    float rs = rsqrtf(var + 1e-5f);
#pragma unroll
    for (int i = 0; i < 8; ++i) {
      int c = lane * 8 + i;
      sX[ml * 256 + c] = (__bf16)((v[i] - mean) * rs * g2[c] + b2[c]);
    }
  }
  __syncthreads();

  // fc1 + exact GELU
  for (int t = 0; t < 32; ++t) {
    int nt = wave * 8 + (t >> 2), mt = t & 3;
    v8f acc = {};
    const __bf16* xa = sX + (mt * 16 + l16) * 256 + lh * 8;
    const __bf16* wb = wfc1T + ((size_t)(nt * 16 + l16)) * 256 + lh * 8;
#pragma unroll
    for (int kt = 0; kt < 8; ++kt) {
      v16bf a  = ld_frag(xa + kt * 32, xa + kt * 32 + 16);
      v16bf bf = ld_frag(wb + kt * 32, wb + kt * 32 + 16);
      acc = __builtin_amdgcn_wmma_f32_16x16x32_bf16(false, a, false, bf, (short)0, acc, false, false);
    }
    int n = nt * 16 + l16;
    float bias = fc1_b[n];
#pragma unroll
    for (int r = 0; r < 8; ++r) {
      int m = mt * 16 + r + 8 * lh;
      float hv = acc[r] + bias;
      float ge = 0.5f * hv * (1.f + erff(hv * 0.70710678118654752f));
      sH[m * 1024 + n] = (__bf16)ge;
    }
  }
  __syncthreads();

  // fc2 + residual
  for (int t = 0; t < 8; ++t) {
    int nt = wave * 2 + (t >> 2), mt = t & 3;
    v8f acc = {};
    const __bf16* ha = sH + (mt * 16 + l16) * 1024 + lh * 8;
    const __bf16* wb = wfc2T + ((size_t)(nt * 16 + l16)) * 1024 + lh * 8;
    for (int kt = 0; kt < 32; ++kt) {
      v16bf a  = ld_frag(ha + kt * 32, ha + kt * 32 + 16);
      v16bf bf = ld_frag(wb + kt * 32, wb + kt * 32 + 16);
      acc = __builtin_amdgcn_wmma_f32_16x16x32_bf16(false, a, false, bf, (short)0, acc, false, false);
    }
    int c = nt * 16 + l16;
    float bias = fc2_b[c];
#pragma unroll
    for (int r = 0; r < 8; ++r) {
      int m = mt * 16 + r + 8 * lh;
      size_t idx = (t0 + m) * 256 + c;
      out[idx] = out[idx] + acc[r] + bias;   // read-before-write by owner thread
    }
  }
}

extern "C" void kernel_launch(void* const* d_in, const int* in_sizes, int n_in,
                              void* d_out, int out_size, void* d_ws, size_t ws_size,
                              hipStream_t stream) {
  (void)in_sizes; (void)n_in; (void)out_size; (void)ws_size;
  const float* x      = (const float*)d_in[0];
  const float* n1g    = (const float*)d_in[1];
  const float* n1b    = (const float*)d_in[2];
  const float* qkv_w  = (const float*)d_in[3];
  const float* qkv_b  = (const float*)d_in[4];
  const float* proj_w = (const float*)d_in[5];
  const float* proj_b = (const float*)d_in[6];
  const float* relb   = (const float*)d_in[7];
  const float* n2g    = (const float*)d_in[8];
  const float* n2b    = (const float*)d_in[9];
  const float* fc1_w  = (const float*)d_in[10];
  const float* fc1_b  = (const float*)d_in[11];
  const float* fc2_w  = (const float*)d_in[12];
  const float* fc2_b  = (const float*)d_in[13];
  float* out = (float*)d_out;

  char* ws = (char*)d_ws;
  __bf16* xw     = (__bf16*)(ws);              // 2048*64*256*2 = 67,108,864 B
  __bf16* wqkvT  = (__bf16*)(ws + 67108864);   // [768][256]
  __bf16* wprojT = (__bf16*)(ws + 67502080);   // [256][256]
  __bf16* wfc1T  = (__bf16*)(ws + 67633152);   // [1024][256]
  __bf16* wfc2T  = (__bf16*)(ws + 68157440);   // [256][1024]

  wconv_kernel<<<(256 * 768 + 255) / 256, 256, 0, stream>>>(qkv_w, wqkvT, 256, 768);
  wconv_kernel<<<(256 * 256 + 255) / 256, 256, 0, stream>>>(proj_w, wprojT, 256, 256);
  wconv_kernel<<<(256 * 1024 + 255) / 256, 256, 0, stream>>>(fc1_w, wfc1T, 256, 1024);
  wconv_kernel<<<(1024 * 256 + 255) / 256, 256, 0, stream>>>(fc2_w, wfc2T, 1024, 256);

  ln1_kernel<<<16384, 256, 0, stream>>>(x, n1g, n1b, xw);
  attn_kernel<<<2048, 256, ATT_LDS, stream>>>(xw, wqkvT, qkv_b, wprojT, proj_b, relb, x, out);
  mlp_kernel<<<2048, 256, MLP_LDS, stream>>>(n2g, n2b, wfc1T, fc1_b, wfc2T, fc2_b, out);
}
